// SemanticCorrespondenceModel_81252191306087
// MI455X (gfx1250) — matrix operand order repaired and checked by reading, not compiled, not run on hardware
//
#include <hip/hip_runtime.h>

// ---------------------------------------------------------------------------
// MI455X / gfx1250 semantic-correspondence pipeline.
//
// GEMM work (~78 GFLOP) runs through v_wmma_f32_16x16x32_bf16 (wave32 WMMA),
// f32 accumulation. K=256 GEMMs (cost volume, proj GEMM) stage their B tile
// in LDS via the Tensor Data Mover (tensor_load_to_lds / s_wait_tensorcnt).
// Geometry: PATCH=14, H=W=518 -> h=w=37, hw=1369 (padded to 1408 = 22*64);
//           K(patch)=588 padded to 608 = 19*32 ; fd=768 ; pd=256
// ---------------------------------------------------------------------------

typedef __bf16 bf16;
typedef __bf16 v16bf __attribute__((ext_vector_type(16)));
typedef __bf16 v8bf  __attribute__((ext_vector_type(8)));
typedef float  v8f   __attribute__((ext_vector_type(8)));
typedef unsigned int u32x4 __attribute__((ext_vector_type(4)));
typedef int          i32x8 __attribute__((ext_vector_type(8)));
typedef int          i32x4 __attribute__((ext_vector_type(4)));

#define HW        1369
#define HWPAD     1408          // 22 * 64
#define GW        37
#define KPATCH    588
#define KPATCH_P  608           // 19 * 32
#define FD        768
#define PD        256
#define NB        16
#define NKP       32
#define IMGW      518
#define MTOT      (2 * NB * HWPAD)   // 45056 rows across {src,trg} x batch

#define HAVE_TDM (__has_builtin(__builtin_amdgcn_tensor_load_to_lds) && \
                  __has_builtin(__builtin_amdgcn_s_wait_tensorcnt))

// ---------------- WMMA fragment helpers (A/B: contiguous K order, shared by both) ----------

// A-matrix 16x32 bf16: lane<16 -> M=lane, K=[k0,k0+16);
//                      lane>=16 -> same M, K=[k0+16,k0+32).
__device__ inline v16bf load_frag_a(const bf16* __restrict__ A, int lda,
                                    int row0, int k0, int lane) {
    int m   = lane & 15;
    int sel = lane >> 4;
    const bf16* p = A + (size_t)(row0 + m) * lda + k0 + sel * 16;
    return *(const v16bf*)(p);          // K = k0+sel*16 .. +15
}

// B-matrix 32x16 bf16 (K x N): lane<16 -> N=lane, K=[k0,k0+16);
//                              lane>=16 -> same N, K=[k0+16,k0+32).
// BT is the K-contiguous (transposed) operand: BT[n][k].
__device__ inline v16bf load_frag_b(const bf16* __restrict__ BT, int ldb,
                                    int col0, int k0, int lane) {
    int n   = lane & 15;
    int sel = lane >> 4;
    const bf16* p = BT + (size_t)(col0 + n) * ldb + k0 + sel * 16;
    return *(const v16bf*)(p);          // 16 contiguous K values (32B)
}

// Same B fragment sourced from an LDS-staged tile (ld fixed = PD): ds_load path.
__device__ inline v16bf load_frag_b_lds(const bf16* Bs, int col0, int k0, int lane) {
    int n   = lane & 15;
    int sel = lane >> 4;
    const bf16* p = Bs + (size_t)(col0 + n) * PD + k0 + sel * 16;
    return *(const v16bf*)(p);
}

__device__ inline v8f wmma_bf16(v16bf a, v16bf b, v8f c) {
    // 8 args: (neg_a, A, neg_b, B, c_mod, C, reuse_a, reuse_b)
    return __builtin_amdgcn_wmma_f32_16x16x32_bf16(
        false, a, false, b, (short)0, c, false, false);
}

// C/D 16x16 f32: VGPR r -> lanes 0-15: M=r, N=lane ; lanes 16-31: M=r+8.
__device__ inline void store_tile(v8f acc, float* __restrict__ Cf,
                                  bf16* __restrict__ Cb, int ldc,
                                  int r0, int c0, int Mlim, int Nlim,
                                  const float* __restrict__ bias,
                                  float scale, int act, int lane) {
    int n    = lane & 15;
    int moff = (lane >> 4) * 8;
    int col  = c0 + n;
    if (col >= Nlim) return;
    float bv = bias ? bias[col] : 0.0f;
#pragma unroll
    for (int r = 0; r < 8; ++r) {
        int row = r0 + moff + r;
        if (row >= Mlim) continue;
        float v = acc[r] * scale + bv;
        if (act == 1)  // exact GELU
            v = 0.5f * v * (1.0f + erff(v * 0.70710678118654752440f));
        if (Cb) Cb[(size_t)row * ldc + col] = (bf16)v;
        else    Cf[(size_t)row * ldc + col] = v;
    }
}

// ---------------- generic WMMA GEMM: C = act(scale * A @ BT^T + bias) -------
// Block = 128 threads = 4 waves; each wave computes a 32x32 tile (2x2 WMMA).
__global__ __launch_bounds__(128) void gemm_wmma_kernel(
    const bf16* __restrict__ A, const bf16* __restrict__ BT,
    float* __restrict__ Cf, bf16* __restrict__ Cb,
    int lda, int ldb, int ldc, int K, int Mlim, int Nlim,
    const float* __restrict__ bias, float scale, int act,
    long long sA, long long sB, long long sC) {
    int z = blockIdx.z;
    A  += (size_t)z * sA;
    BT += (size_t)z * sB;
    if (Cf) Cf += (size_t)z * sC;
    if (Cb) Cb += (size_t)z * sC;

    int lane = threadIdx.x & 31;
    int wid  = threadIdx.x >> 5;
    int r0 = blockIdx.x * 64 + (wid >> 1) * 32;
    int c0 = blockIdx.y * 64 + (wid & 1) * 32;

    v8f a00 = {0,0,0,0,0,0,0,0}, a01 = {0,0,0,0,0,0,0,0};
    v8f a10 = {0,0,0,0,0,0,0,0}, a11 = {0,0,0,0,0,0,0,0};

    for (int k0 = 0; k0 < K; k0 += 32) {
        v16bf fa0 = load_frag_a(A, lda, r0,      k0, lane);
        v16bf fa1 = load_frag_a(A, lda, r0 + 16, k0, lane);
        v16bf fb0 = load_frag_b(BT, ldb, c0,      k0, lane);
        v16bf fb1 = load_frag_b(BT, ldb, c0 + 16, k0, lane);
        if (k0 + 32 < K)  // global_prefetch for next K step of A
            __builtin_prefetch(A + (size_t)(r0 + (lane & 15)) * lda + k0 + 32, 0, 1);
        a00 = wmma_bf16(fa0, fb0, a00);
        a01 = wmma_bf16(fa0, fb1, a01);
        a10 = wmma_bf16(fa1, fb0, a10);
        a11 = wmma_bf16(fa1, fb1, a11);
    }
    store_tile(a00, Cf, Cb, ldc, r0,      c0,      Mlim, Nlim, bias, scale, act, lane);
    store_tile(a01, Cf, Cb, ldc, r0,      c0 + 16, Mlim, Nlim, bias, scale, act, lane);
    store_tile(a10, Cf, Cb, ldc, r0 + 16, c0,      Mlim, Nlim, bias, scale, act, lane);
    store_tile(a11, Cf, Cb, ldc, r0 + 16, c0 + 16, Mlim, Nlim, bias, scale, act, lane);
}

// ---------------- K=256 WMMA GEMM with TDM-staged B tile in LDS -------------
// Per block: the 64x256 bf16 B tile (32 KB) is DMA'd once into LDS by the
// Tensor Data Mover (TENSORcnt-tracked), then all 4 waves run the fully
// unrolled K loop with B fragments from LDS (ds_load) and A from global.
__global__ __launch_bounds__(128) void gemm_wmma_tdm_kernel(
    const bf16* __restrict__ A, const bf16* __restrict__ BT,
    float* __restrict__ Cf, bf16* __restrict__ Cb,
    int lda, int ldc, int Mlim, int Nlim,
    const float* __restrict__ bias, float scale, int act,
    long long sA, long long sB, long long sC) {
    __shared__ bf16 Bs[64 * PD];    // 32 KB

    int z = blockIdx.z;
    A  += (size_t)z * sA;
    BT += (size_t)z * sB;
    if (Cf) Cf += (size_t)z * sC;
    if (Cb) Cb += (size_t)z * sC;

    int lane = threadIdx.x & 31;
    int wid  = threadIdx.x >> 5;
    int cblk = blockIdx.y * 64;

#if HAVE_TDM
    if (threadIdx.x < 32) {  // wave 0 issues the tensor DMA (EXEC-independent)
        unsigned long long ga =
            (unsigned long long)(size_t)(BT + (size_t)cblk * PD);
        unsigned lds_off = (unsigned)(unsigned long long)(size_t)&Bs[0];
        // D# group 0: count=1 | lds_addr | global_addr | type=2
        u32x4 g0 = { 1u,
                     lds_off,
                     (unsigned)(ga & 0xffffffffu),
                     (unsigned)((ga >> 32) & 0x1ffffffu) | (2u << 30) };
        // D# group 1: data_size=2B; tensor 256x64, tile 256x64, stride0=256
        i32x8 g1 = { 0x00010000,            // data_size=1 (2 bytes)
                     (int)(256u << 16),     // tensor_dim0[15:0]=256
                     (int)(64u  << 16),     // tensor_dim1[15:0]=64
                     (int)(256u << 16),     // tile_dim0=256
                     64,                    // tile_dim1=64, tile_dim2=0
                     256,                   // tensor_dim0_stride=256
                     0, 0 };
        i32x4 gz4 = { 0, 0, 0, 0 };         // groups 2/3 unused (<=2D tensor)
        i32x8 gz8 = { 0, 0, 0, 0, 0, 0, 0, 0 };
        // 6-arg toolchain variant: (v4u, v8i, v4i, v4i, v8i, cpol)
        __builtin_amdgcn_tensor_load_to_lds(g0, g1, gz4, gz4, gz8, 0);
        __builtin_amdgcn_s_wait_tensorcnt(0);
    }
#else
    for (int i = threadIdx.x; i < 64 * PD; i += 128)
        Bs[i] = BT[(size_t)cblk * PD + i];   // rows contiguous: ldb == PD
#endif
    __syncthreads();

    int r0  = blockIdx.x * 64 + (wid >> 1) * 32;
    int c0l = (wid & 1) * 32;                // column base inside LDS tile
    int c0  = cblk + c0l;

    v8f a00 = {0,0,0,0,0,0,0,0}, a01 = {0,0,0,0,0,0,0,0};
    v8f a10 = {0,0,0,0,0,0,0,0}, a11 = {0,0,0,0,0,0,0,0};

#pragma unroll
    for (int k0 = 0; k0 < PD; k0 += 32) {
        v16bf fa0 = load_frag_a(A, lda, r0,      k0, lane);
        v16bf fa1 = load_frag_a(A, lda, r0 + 16, k0, lane);
        v16bf fb0 = load_frag_b_lds(Bs, c0l,      k0, lane);
        v16bf fb1 = load_frag_b_lds(Bs, c0l + 16, k0, lane);
        a00 = wmma_bf16(fa0, fb0, a00);
        a01 = wmma_bf16(fa0, fb1, a01);
        a10 = wmma_bf16(fa1, fb0, a10);
        a11 = wmma_bf16(fa1, fb1, a11);
    }
    store_tile(a00, Cf, Cb, ldc, r0,      c0,      Mlim, Nlim, bias, scale, act, lane);
    store_tile(a01, Cf, Cb, ldc, r0,      c0 + 16, Mlim, Nlim, bias, scale, act, lane);
    store_tile(a10, Cf, Cb, ldc, r0 + 16, c0,      Mlim, Nlim, bias, scale, act, lane);
    store_tile(a11, Cf, Cb, ldc, r0 + 16, c0 + 16, Mlim, Nlim, bias, scale, act, lane);
}

// ---------------- weight prep: transpose f32 -> bf16, K-contiguous ----------
__global__ void prep_weights_kernel(const float* __restrict__ Wp,
                                    const float* __restrict__ W1,
                                    const float* __restrict__ W2,
                                    bf16* __restrict__ WpT,
                                    bf16* __restrict__ W1T,
                                    bf16* __restrict__ W2T) {
    int idx = blockIdx.x * 256 + threadIdx.x;
    if (idx < FD * KPATCH_P) {                       // WpT[768][608]
        int n = idx / KPATCH_P, k = idx % KPATCH_P;
        WpT[idx] = (bf16)((k < KPATCH) ? Wp[(size_t)k * FD + n] : 0.0f);
        return;
    }
    idx -= FD * KPATCH_P;
    if (idx < PD * FD) {                             // W1T[256][768]
        int n = idx / FD, k = idx % FD;
        W1T[idx] = (bf16)W1[(size_t)k * PD + n];
        return;
    }
    idx -= PD * FD;
    if (idx < PD * PD) {                             // W2T[256][256]
        int n = idx / PD, k = idx % PD;
        W2T[idx] = (bf16)W2[(size_t)k * PD + n];
    }
}

// ---------------- patchify both images -> bf16 rows, zero-padded ------------
__global__ void patchify_kernel(const float* __restrict__ src,
                                const float* __restrict__ trg,
                                bf16* __restrict__ P) {
    size_t idx = (size_t)blockIdx.x * 256 + threadIdx.x;
    const size_t total = (size_t)MTOT * KPATCH_P;
    if (idx >= total) return;
    int k = (int)(idx % KPATCH_P);
    size_t t = idx / KPATCH_P;
    int row = (int)(t % HWPAD); t /= HWPAD;
    int b   = (int)(t % NB);
    int im  = (int)(t / NB);
    float v = 0.0f;
    if (row < HW && k < KPATCH) {
        int py = row / GW, px = row % GW;
        int c = k / 196, rem = k % 196, i = rem / 14, j = rem % 14;
        const float* img = (im == 0) ? src : trg;
        v = img[(((size_t)b * 3 + c) * IMGW + (py * 14 + i)) * IMGW + (px * 14 + j)];
    }
    P[idx] = (bf16)v;
}

// ---------------- row L2 normalize: f32 z -> bf16 features ------------------
__global__ void normalize_kernel(const float* __restrict__ Z, bf16* __restrict__ F) {
    int row = blockIdx.x, tid = threadIdx.x;
    __shared__ float red[256];
    float x = Z[(size_t)row * PD + tid];
    red[tid] = x * x;
    __syncthreads();
    for (int s = 128; s > 0; s >>= 1) {
        if (tid < s) red[tid] += red[tid + s];
        __syncthreads();
    }
    float inv = 1.0f / (sqrtf(red[0]) + 1e-12f);
    F[(size_t)row * PD + tid] = (bf16)(x * inv);
}

// ---------------- adaptive windowed soft-argmax per keypoint ----------------
__global__ void keypoint_kernel(const float* __restrict__ cost,
                                const float* __restrict__ src_kps,
                                float* __restrict__ pred,
                                float* __restrict__ entv) {
    int n = blockIdx.x, b = blockIdx.y, tid = threadIdx.x;
    __shared__ float red_f[256], red_g[256], red_h[256];
    __shared__ int   red_i[256];
    __shared__ float s_gmax, s_S1, s_S2;
    __shared__ int   s_best;

    float kxf = src_kps[((size_t)b * NKP + n) * 2 + 0];
    float kyf = src_kps[((size_t)b * NKP + n) * 2 + 1];
    int kx = min(max((int)(kxf / (float)IMGW * (float)GW), 0), GW - 1);
    int ky = min(max((int)(kyf / (float)IMGW * (float)GW), 0), GW - 1);
    const float* row = cost + ((size_t)b * HW + (size_t)(ky * GW + kx)) * HW;

    // 1) max + argmax (first-max tiebreak)
    float mv = -3.0e38f; int mi = 0;
    for (int t = tid; t < HW; t += 256) {
        float v = row[t];
        if (v > mv) { mv = v; mi = t; }
    }
    red_f[tid] = mv; red_i[tid] = mi;
    __syncthreads();
    for (int s = 128; s > 0; s >>= 1) {
        if (tid < s) {
            float v2 = red_f[tid + s]; int i2 = red_i[tid + s];
            if (v2 > red_f[tid] || (v2 == red_f[tid] && i2 < red_i[tid])) {
                red_f[tid] = v2; red_i[tid] = i2;
            }
        }
        __syncthreads();
    }
    if (tid == 0) { s_gmax = red_f[0]; s_best = red_i[0]; }
    __syncthreads();
    float gmax = s_gmax; int best = s_best;

    // 2) logsumexp + entropy: ent = log(S1) - S2/S1 with x' = x - max
    float s1 = 0.0f, s2 = 0.0f;
    for (int t = tid; t < HW; t += 256) {
        float x = row[t] - gmax, e = expf(x);
        s1 += e; s2 += x * e;
    }
    red_f[tid] = s1; red_g[tid] = s2;
    __syncthreads();
    for (int s = 128; s > 0; s >>= 1) {
        if (tid < s) { red_f[tid] += red_f[tid + s]; red_g[tid] += red_g[tid + s]; }
        __syncthreads();
    }
    if (tid == 0) { s_S1 = red_f[0]; s_S2 = red_g[0]; }
    __syncthreads();
    float ent = logf(s_S1) - s_S2 / s_S1;

    // 3) entropy-driven radius, windowed softmax centroid
    float rr = rintf(2.0f + 5.0f * ent / logf((float)HW));
    rr = fminf(fmaxf(rr, 2.0f), 7.0f);
    float cx = (float)(best % GW), cy = (float)(best / GW);
    float sw = 0.0f, sx = 0.0f, sy = 0.0f;
    for (int t = tid; t < HW; t += 256) {
        float xs = (float)(t % GW), ys = (float)(t / GW);
        if (fabsf(xs - cx) <= rr && fabsf(ys - cy) <= rr) {
            float e = expf(row[t] - gmax);   // window max == global max
            sw += e; sx += e * xs; sy += e * ys;
        }
    }
    red_f[tid] = sw; red_g[tid] = sx; red_h[tid] = sy;
    __syncthreads();
    for (int s = 128; s > 0; s >>= 1) {
        if (tid < s) {
            red_f[tid] += red_f[tid + s];
            red_g[tid] += red_g[tid + s];
            red_h[tid] += red_h[tid + s];
        }
        __syncthreads();
    }
    if (tid == 0) {
        float gx = red_g[0] / red_f[0], gy = red_h[0] / red_f[0];
        pred[((size_t)b * NKP + n) * 2 + 0] = (gx + 0.5f) / (float)GW * (float)IMGW;
        pred[((size_t)b * NKP + n) * 2 + 1] = (gy + 0.5f) / (float)GW * (float)IMGW;
        entv[(size_t)b * NKP + n] = ent;
    }
}

// ---------------------------------------------------------------------------
extern "C" void kernel_launch(void* const* d_in, const int* in_sizes, int n_in,
                              void* d_out, int out_size, void* d_ws, size_t ws_size,
                              hipStream_t stream) {
    (void)in_sizes; (void)n_in; (void)out_size; (void)ws_size;
    const float* src_img = (const float*)d_in[0];
    const float* trg_img = (const float*)d_in[1];
    const float* src_kps = (const float*)d_in[2];
    const float* W_patch = (const float*)d_in[3];
    const float* b_patch = (const float*)d_in[4];
    const float* W1      = (const float*)d_in[5];
    const float* b1      = (const float*)d_in[6];
    const float* W2      = (const float*)d_in[7];
    const float* b2      = (const float*)d_in[8];

    float* cost = (float*)d_out;                              // (16,1369,1369)
    float* pred = cost + (size_t)NB * HW * HW;                // (16,32,2)
    float* entv = pred + (size_t)NB * NKP * 2;                // (16,32)

    // ---- workspace layout (all 256B aligned); ~148.5 MB total ----
    char* ws = (char*)d_ws;
    const size_t o_WpT  = 0;                                      // 768*608 bf16
    const size_t o_W1T  = o_WpT + (size_t)FD * KPATCH_P * 2;      // 256*768 bf16
    const size_t o_W2T  = o_W1T + (size_t)PD * FD * 2;            // 256*256 bf16
    const size_t o_P    = o_W2T + (size_t)PD * PD * 2;            // 45056*608 bf16 (reused: f32 z)
    const size_t o_feat = o_P   + (size_t)MTOT * KPATCH_P * 2;    // 45056*768 bf16 (reused: featN)
    const size_t o_z1   = o_feat + (size_t)MTOT * FD * 2;         // 45056*256 bf16

    bf16*  WpT   = (bf16*)(ws + o_WpT);
    bf16*  W1T   = (bf16*)(ws + o_W1T);
    bf16*  W2T   = (bf16*)(ws + o_W2T);
    bf16*  P     = (bf16*)(ws + o_P);
    float* Zf    = (float*)(ws + o_P);      // reuse: P dead after GEMM1
    bf16*  feat  = (bf16*)(ws + o_feat);
    bf16*  featN = (bf16*)(ws + o_feat);    // reuse: feat dead after GEMM2
    bf16*  z1    = (bf16*)(ws + o_z1);

    // 1) weights -> bf16 K-contiguous
    prep_weights_kernel<<<(FD * KPATCH_P + PD * FD + PD * PD) / 256, 256, 0, stream>>>(
        W_patch, W1, W2, WpT, W1T, W2T);

    // 2) patchify both images (zero-padded K and M)
    {
        size_t total = (size_t)MTOT * KPATCH_P;
        patchify_kernel<<<(unsigned)((total + 255) / 256), 256, 0, stream>>>(src_img, trg_img, P);
    }

    // 3) feat = P @ Wp + b_patch          (M=45056, K=608, N=768) -> bf16
    gemm_wmma_kernel<<<dim3(MTOT / 64, FD / 64, 1), 128, 0, stream>>>(
        P, WpT, nullptr, feat, KPATCH_P, KPATCH_P, FD, KPATCH_P, MTOT, FD,
        b_patch, 1.0f, 0, 0, 0, 0);

    // 4) z1 = gelu(feat @ W1 + b1)        (K=768, N=256) -> bf16
    gemm_wmma_kernel<<<dim3(MTOT / 64, PD / 64, 1), 128, 0, stream>>>(
        feat, W1T, nullptr, z1, FD, FD, PD, FD, MTOT, PD,
        b1, 1.0f, 1, 0, 0, 0);

    // 5) z  = z1 @ W2 + b2                (K=256, N=256) -> f32 (into P region)
    //    TDM-staged B tile (W2T block) in LDS.
    gemm_wmma_tdm_kernel<<<dim3(MTOT / 64, PD / 64, 1), 128, 0, stream>>>(
        z1, W2T, Zf, nullptr, PD, PD, MTOT, PD,
        b2, 1.0f, 0, 0, 0, 0);

    // 6) featN = z / (||z|| + 1e-12) -> bf16 (into feat region)
    normalize_kernel<<<MTOT, 256, 0, stream>>>(Zf, featN);

    // 7) cost volume: per batch, srcF @ trgF^T * 20  (M=N=1369, K=256) -> f32
    //    TDM-staged trg-feature tile in LDS.
    gemm_wmma_tdm_kernel<<<dim3(HWPAD / 64, HWPAD / 64, NB), 128, 0, stream>>>(
        featN, featN + (size_t)NB * HWPAD * PD, cost, nullptr,
        PD, HW, HW, HW, nullptr, 20.0f, 0,
        (long long)HWPAD * PD, (long long)HWPAD * PD, (long long)HW * HW);

    // 8) keypoints: entropy-adaptive windowed soft-argmax
    keypoint_kernel<<<dim3(NKP, NB), 256, 0, stream>>>(cost, src_kps, pred, entv);
}
